// AttentionHead_2130303779020
// MI455X (gfx1250) — matrix operand order repaired and checked by reading, not compiled, not run on hardware
//
#include <hip/hip_runtime.h>
#include <hip/hip_bf16.h>

#define D_EMB  1024
#define D_HEAD 64
#define NBATCH 4
#define SEQT   4096
#define NROW   (NBATCH * SEQT)   // 16384 total rows

typedef _Float16 v4h  __attribute__((ext_vector_type(4)));
typedef _Float16 v8h  __attribute__((ext_vector_type(8)));
typedef _Float16 v16h __attribute__((ext_vector_type(16)));
typedef float    v8f  __attribute__((ext_vector_type(8)));

static __device__ __forceinline__ v8f wmma_f16(v16h a, v16h b, v8f c) {
  // 8 args: (neg_a, A, neg_b, B, c_mod, C, reuse_a, reuse_b)
  return __builtin_amdgcn_wmma_f32_16x16x32_f16(false, a, false, b, (short)0, c,
                                                false, false);
}

static __device__ __forceinline__ v16h pack16(v8h lo, v8h hi) {
  return __builtin_shufflevector(lo, hi, 0, 1, 2, 3, 4, 5, 6, 7,
                                 8, 9, 10, 11, 12, 13, 14, 15);
}

// ---------------------------------------------------------------------------
// Kernel 0: one-time f32 -> f16 conversion of a 64x1024 weight matrix.
// 4 elements per thread, packed cvt + 8B store.  64 blocks x 256 threads.
// ---------------------------------------------------------------------------
__global__ __launch_bounds__(256) void wcvt_kernel(const float* __restrict__ w,
                                                   _Float16* __restrict__ wh) {
  const int i = (blockIdx.x * 256 + threadIdx.x) * 4;
  const float4 f = *(const float4*)(w + i);
  v4h o;
  o[0] = (_Float16)f.x; o[1] = (_Float16)f.y;
  o[2] = (_Float16)f.z; o[3] = (_Float16)f.w;
  *(v4h*)(wh + i) = o;
}

// ---------------------------------------------------------------------------
// Kernel 1: fused q/k/v projection.  GEMM (16384 x 1024) x (1024 x 64) three
// times.  One wave per 16-row tile; x converted f32->f16 on the fly, weights
// already f16 (L2-resident after first touch).
// q,k stored row-major f16 [row][64]; v stored transposed f16 [b][64][T].
// ---------------------------------------------------------------------------
__global__ __launch_bounds__(128) void qkv_proj_kernel(
    const float*    __restrict__ x,
    const _Float16* __restrict__ Wkh,
    const _Float16* __restrict__ Wqh,
    const _Float16* __restrict__ Wvh,
    _Float16* __restrict__ qh, _Float16* __restrict__ kh,
    _Float16* __restrict__ vT) {
  const int lane = threadIdx.x & 31;
  const int wave = threadIdx.x >> 5;
  const int g    = lane >> 4;     // lane group (0: lanes 0-15, 1: lanes 16-31)
  const int mrow = lane & 15;
  const int rowBase = (blockIdx.x * 4 + wave) * 16;

  const v8f Z = {0.f, 0.f, 0.f, 0.f, 0.f, 0.f, 0.f, 0.f};
  v8f Cq[4], Ck[4], Cv[4];
#pragma unroll
  for (int nf = 0; nf < 4; ++nf) { Cq[nf] = Z; Ck[nf] = Z; Cv[nf] = Z; }

  const float* xrow = x + (size_t)(rowBase + mrow) * D_EMB;

  for (int chunk = 0; chunk < D_EMB / 32; ++chunk) {
    // A fragment: lane L row mrow, e0..15 -> k=g*16+e
    const float* xp = xrow + chunk * 32;
    v8f xa = *(const v8f*)(xp + g * 16);
    v8f xb = *(const v8f*)(xp + g * 16 + 8);
    v16h A;
#pragma unroll
    for (int i = 0; i < 8; ++i) {
      A[i]     = (_Float16)xa[i];
      A[8 + i] = (_Float16)xb[i];
    }

#pragma unroll
    for (int nf = 0; nf < 4; ++nf) {
      // B fragment: lane n col = nf*16+mrow (W output row), k = g*16+e (emb)
      const size_t woff = (size_t)(nf * 16 + mrow) * D_EMB + chunk * 32 + g * 16;

      v16h Bq = pack16(*(const v8h*)(Wqh + woff), *(const v8h*)(Wqh + woff + 8));
      Cq[nf] = wmma_f16(A, Bq, Cq[nf]);

      v16h Bk = pack16(*(const v8h*)(Wkh + woff), *(const v8h*)(Wkh + woff + 8));
      Ck[nf] = wmma_f16(A, Bk, Ck[nf]);

      v16h Bv = pack16(*(const v8h*)(Wvh + woff), *(const v8h*)(Wvh + woff + 8));
      Cv[nf] = wmma_f16(A, Bv, Cv[nf]);
    }
  }

  // C/D layout: VGPR r holds (M = r + 8*g, N = mrow)
#pragma unroll
  for (int nf = 0; nf < 4; ++nf) {
#pragma unroll
    for (int r = 0; r < 8; ++r) {
      const int row = rowBase + r + 8 * g;
      const int col = nf * 16 + mrow;
      qh[(size_t)row * D_HEAD + col] = (_Float16)Cq[nf][r];
      kh[(size_t)row * D_HEAD + col] = (_Float16)Ck[nf][r];
      const int b = row >> 12;           // / SEQT
      const int t = row & (SEQT - 1);
      vT[((size_t)b * D_HEAD + col) * SEQT + t] = (_Float16)Cv[nf][r];
    }
  }
}

// ---------------------------------------------------------------------------
// Kernel 2: flash attention (causal, online softmax).  One wave per 16-query
// tile, 32 keys per iteration (K dim of the PV matmul = 32).
// ---------------------------------------------------------------------------
__global__ __launch_bounds__(128) void flash_attn_kernel(
    const _Float16* __restrict__ qh, const _Float16* __restrict__ kh,
    const _Float16* __restrict__ vT, float* __restrict__ out) {
  __shared__ _Float16 pbuf[4][16][32];   // per-wave P tile (C-layout -> A-layout bounce)

  const int lane = threadIdx.x & 31;
  const int wave = threadIdx.x >> 5;
  const int g    = lane >> 4;
  const int mrow = lane & 15;
  const int rowBase = (blockIdx.x * 4 + wave) * 16;
  const int b     = rowBase >> 12;
  const int qbase = rowBase & (SEQT - 1);

  const _Float16* Qb = qh + (size_t)b * SEQT * D_HEAD;
  const _Float16* Kb = kh + (size_t)b * SEQT * D_HEAD;
  const _Float16* Vb = vT + (size_t)b * D_HEAD * SEQT;

  // Q A-fragments, pre-scaled by 1/sqrt(64) = 0.125 (exact in f16)
  v16h QA[2];
#pragma unroll
  for (int c = 0; c < 2; ++c) {
    const _Float16* qrow = Qb + (size_t)(qbase + mrow) * D_HEAD + c * 32;
    v16h A = pack16(*(const v8h*)(qrow + g * 16),
                    *(const v8h*)(qrow + g * 16 + 8));
#pragma unroll
    for (int i = 0; i < 16; ++i) A[i] = A[i] * (_Float16)0.125f;
    QA[c] = A;
  }

  const v8f Z = {0.f, 0.f, 0.f, 0.f, 0.f, 0.f, 0.f, 0.f};
  v8f O[4];
#pragma unroll
  for (int nf = 0; nf < 4; ++nf) O[nf] = Z;
  float mr[8], lr[8];
#pragma unroll
  for (int r = 0; r < 8; ++r) { mr[r] = -__builtin_inff(); lr[r] = 0.f; }

  const int kend = qbase + 16;           // exclusive causal key bound
  for (int kb = 0; kb < kend; kb += 32) {
    // S = Q (16x64) * K^T (64x32): 2 N-fragments x 2 K-chunks
    v8f S[2] = {Z, Z};
#pragma unroll
    for (int c = 0; c < 2; ++c) {
#pragma unroll
      for (int j = 0; j < 2; ++j) {
        const _Float16* krow =
            Kb + (size_t)(kb + j * 16 + mrow) * D_HEAD + c * 32 + g * 16;
        v16h Bf = pack16(*(const v8h*)krow, *(const v8h*)(krow + 8));
        S[j] = wmma_f16(QA[c], Bf, S[j]);
      }
    }

    // causal mask + online softmax (rows live across 16-lane groups)
#pragma unroll
    for (int r = 0; r < 8; ++r) {
      const int qi = qbase + r + 8 * g;
      float s0 = (kb + mrow      > qi) ? -__builtin_inff() : S[0][r];
      float s1 = (kb + 16 + mrow > qi) ? -__builtin_inff() : S[1][r];
      float mn = fmaxf(mr[r], fmaxf(s0, s1));
      mn = fmaxf(mn, __shfl_xor(mn, 1, 32));
      mn = fmaxf(mn, __shfl_xor(mn, 2, 32));
      mn = fmaxf(mn, __shfl_xor(mn, 4, 32));
      mn = fmaxf(mn, __shfl_xor(mn, 8, 32));
      const float p0 = __expf(s0 - mn);
      const float p1 = __expf(s1 - mn);
      const float alpha = __expf(mr[r] - mn);
      mr[r] = mn;
      float rs = p0 + p1;
      rs += __shfl_xor(rs, 1, 32);
      rs += __shfl_xor(rs, 2, 32);
      rs += __shfl_xor(rs, 4, 32);
      rs += __shfl_xor(rs, 8, 32);
      lr[r] = lr[r] * alpha + rs;
#pragma unroll
      for (int nf = 0; nf < 4; ++nf) O[nf][r] *= alpha;
      pbuf[wave][r + 8 * g][mrow]      = (_Float16)p0;
      pbuf[wave][r + 8 * g][mrow + 16] = (_Float16)p1;
    }

    // DS ops from one wave are in-order; wait for the P stores, keep the
    // compiler from reordering the re-gather above them.
    asm volatile("s_wait_dscnt 0" ::: "memory");

    // Re-gather P in A-fragment layout (16x32, contraction dim = 32 keys)
    const _Float16* prow = &pbuf[wave][mrow][0];
    v16h PA = pack16(*(const v8h*)(prow + g * 16),
                     *(const v8h*)(prow + g * 16 + 8));

    // O += P * V ; B-fragments are contiguous rows of vT
#pragma unroll
    for (int nf = 0; nf < 4; ++nf) {
      const _Float16* vrow = Vb + (size_t)(nf * 16 + mrow) * SEQT + kb + g * 16;
      v16h Bv = pack16(*(const v8h*)vrow, *(const v8h*)(vrow + 8));
      O[nf] = wmma_f16(PA, Bv, O[nf]);
    }
  }

  float inv[8];
#pragma unroll
  for (int r = 0; r < 8; ++r) inv[r] = 1.f / lr[r];
#pragma unroll
  for (int nf = 0; nf < 4; ++nf) {
#pragma unroll
    for (int r = 0; r < 8; ++r) {
      out[(size_t)(rowBase + r + 8 * g) * D_HEAD + nf * 16 + mrow] =
          O[nf][r] * inv[r];
    }
  }
}

// ---------------------------------------------------------------------------
extern "C" void kernel_launch(void* const* d_in, const int* in_sizes, int n_in,
                              void* d_out, int out_size, void* d_ws,
                              size_t ws_size, hipStream_t stream) {
  (void)in_sizes; (void)n_in; (void)out_size; (void)ws_size;
  const float* x  = (const float*)d_in[0];
  const float* Wk = (const float*)d_in[1];
  const float* Wq = (const float*)d_in[2];
  const float* Wv = (const float*)d_in[3];
  float* out = (float*)d_out;

  // workspace layout (f16): q | k | vT | Wk | Wq | Wv
  _Float16* qh  = (_Float16*)d_ws;
  _Float16* kh  = qh + (size_t)NROW * D_HEAD;
  _Float16* vT  = kh + (size_t)NROW * D_HEAD;
  _Float16* Wkh = vT + (size_t)NROW * D_HEAD;
  _Float16* Wqh = Wkh + (size_t)D_HEAD * D_EMB;
  _Float16* Wvh = Wqh + (size_t)D_HEAD * D_EMB;

  // one-time weight conversion: 64x1024 f32 -> f16 each
  wcvt_kernel<<<dim3(D_HEAD * D_EMB / 1024), dim3(256), 0, stream>>>(Wk, Wkh);
  wcvt_kernel<<<dim3(D_HEAD * D_EMB / 1024), dim3(256), 0, stream>>>(Wq, Wqh);
  wcvt_kernel<<<dim3(D_HEAD * D_EMB / 1024), dim3(256), 0, stream>>>(Wv, Wvh);

  dim3 block(128);                 // 4 waves (wave32) per workgroup
  dim3 grid(NROW / 64);            // one 16-row tile per wave -> 256 blocks

  qkv_proj_kernel<<<grid, block, 0, stream>>>(x, Wkh, Wqh, Wvh, qh, kh, vT);
  flash_attn_kernel<<<grid, block, 0, stream>>>(qh, kh, vT, out);
}
